// LocalPairLayer_49744311222487
// MI455X (gfx1250) — compile-verified, matrix operand order checked by reading, not run on hardware
//
#include <hip/hip_runtime.h>

typedef unsigned short u16;
typedef __attribute__((ext_vector_type(16))) __bf16 v16bf;
typedef __attribute__((ext_vector_type(8)))  float  v8f;
typedef __attribute__((ext_vector_type(8)))  u16    u16x8;

#define THREADS 256
#define BT   16        // batches per workgroup
#define ROWS 96        // 6 pairs * BT, row = p*16 + b

// ---- workspace (bf16 transposed weights), element offsets ----
#define OFF_PROJ 0L          // 256x512
#define OFF_WQ   131072L     // 2 x 256x256
#define OFF_WK   262144L
#define OFF_WV   393216L
#define OFF_WO   524288L
#define OFF_F1   655360L     // 2 x 1024x256
#define OFF_F2   1179648L    // 2 x 256x1024
#define OFF_WE   1703936L    // 3 x 256x256
#define OFF_PE1  1900544L
#define OFF_PE2  2097152L
#define OFF_WR   2293760L
#define OFF_PR1  2490368L
#define OFF_PR2  2686976L

// ---- dynamic LDS layout (bytes) ----
#define SM_PAIR 0          // f32 96x256   (98304)
#define SM_ABF  98304      // bf16 96x256  (49152)
#define SM_R3   147456     // 147456: sIn 96x512 bf16 | sQ/sK/sV bf16 | msgE f32
#define SM_MISC 294912     // eq/ek/ev tables + valid flags
#define SM_TOTAL 297216

__constant__ int c_lab[36] = {
  0,1,2,2,2,2,
  1,0,3,3,3,3,
  2,3,0,4,4,4,
  2,3,4,0,4,4,
  2,3,4,4,0,4,
  2,3,4,4,4,0};
__constant__ int c_roleB[6]    = {0, 65536, 131072, 131072, 131072, 131072};
__constant__ int c_roleBias[6] = {0, 256, 512, 512, 512, 512};

static __device__ __forceinline__ u16 f2bf(float f) {
  union { float f; unsigned u; } x; x.f = f;
  unsigned r = x.u + 0x7FFFu + ((x.u >> 16) & 1u);
  return (u16)(r >> 16);
}
static __device__ __forceinline__ float bf2f(u16 h) {
  union { unsigned u; float f; } x; x.u = ((unsigned)h) << 16;
  return x.f;
}
static __device__ __forceinline__ float gelu_f(float x) {
  return 0.5f * x * (1.0f + erff(x * 0.70710678118654752f));
}

union Frag { u16x8 h[2]; v16bf v; };

// 16x32 bf16 operand fragment, rows striped per documented CDNA5 layout.
// base -> element (row0, k0) of row-major matrix with leading dim ld.
static __device__ __forceinline__ v16bf load_frag(const u16* base, int ld) {
  int lane = threadIdx.x & 31;
  int r = lane & 15;
  int c = (lane >> 4) * 8;
  Frag f;
  const u16* p = base + r * ld + c;
  f.h[0] = *reinterpret_cast<const u16x8*>(p);
  f.h[1] = *reinterpret_cast<const u16x8*>(p + 16);
  return f.v;
}

// Generic WMMA GEMM: A (LDS bf16, M=Mt*16 x K), B (global bf16, N-major: WT[n][k]),
// out ld fixed 256. Epilogue: bias + act (0 none / 1 gelu / 2 relu),
// optional f32 store-or-accumulate and/or bf16 store.
static __device__ __forceinline__ void gemm(
    const u16* A, int lda, const u16* B, int ldb,
    const float* bias, const int* mtBoff, const int* mtBiasOff,
    float* outF, bool accum, u16* outH, int act, int Mt, int Ksteps)
{
  int wave = threadIdx.x >> 5;
  int lane = threadIdx.x & 31;
  int total = Mt * 16;
  for (int t = wave; t < total; t += 8) {
    int mt = t >> 4, nt = t & 15;
    const u16* Ab = A + mt * 16 * lda;
    const u16* Bb = B + (mtBoff ? mtBoff[mt] : 0) + nt * 16 * ldb;
    v8f acc = {0.f,0.f,0.f,0.f,0.f,0.f,0.f,0.f};
    for (int ks = 0; ks < Ksteps; ks++) {
      v16bf av = load_frag(Ab + ks * 32, lda);
      v16bf bv = load_frag(Bb + ks * 32, ldb);
      acc = __builtin_amdgcn_wmma_f32_16x16x32_bf16(
          false, av, false, bv, (short)0, acc, false, false);
    }
    int col = (nt << 4) + (lane & 15);
    float bvs = bias ? bias[(mtBiasOff ? mtBiasOff[mt] : 0) + col] : 0.f;
    int r0 = (mt << 4) + ((lane >> 4) << 3);
    for (int j = 0; j < 8; j++) {
      float v = acc[j] + bvs;
      if (act == 1) v = gelu_f(v);
      else if (act == 2) v = fmaxf(v, 0.f);
      int idx = (r0 + j) * 256 + col;
      if (outF) { if (accum) outF[idx] += v; else outF[idx] = v; }
      if (outH) outH[idx] = f2bf(v);
    }
  }
}

// Row-wise LayerNorm over 96 rows x 256 cols, wave32 shuffle reduction.
static __device__ __forceinline__ void ln_rows(
    float* buf, u16* outH, const float* s, const float* bsh, const int* validArr)
{
  int wave = threadIdx.x >> 5, lane = threadIdx.x & 31;
  for (int r = wave; r < ROWS; r += 8) {
    float v[8]; float sum = 0.f, sq = 0.f;
    for (int k = 0; k < 8; k++) {
      v[k] = buf[r * 256 + lane + 32 * k];
      sum += v[k]; sq += v[k] * v[k];
    }
    for (int m = 16; m >= 1; m >>= 1) {
      sum += __shfl_xor(sum, m, 32);
      sq  += __shfl_xor(sq,  m, 32);
    }
    float mu = sum * (1.f / 256.f);
    float var = sq * (1.f / 256.f) - mu * mu;
    float rs = rsqrtf(var + 1e-5f);
    float msk = 1.f;
    if (validArr) { int p = r >> 4, b = r & 15; msk = validArr[b * 6 + p] ? 1.f : 0.f; }
    for (int k = 0; k < 8; k++) {
      int c = lane + 32 * k;
      float o = ((v[k] - mu) * rs * s[c] + bsh[c]) * msk;
      buf[r * 256 + c] = o;
      if (outH) outH[r * 256 + c] = f2bf(o);
    }
  }
}

struct KParams {
  const float* x; const unsigned char* mask; const u16* ws;
  const float* proj_b;
  const float *bq, *bk, *bv, *bo, *ln1_s, *ln1_b, *ffn_b1, *ffn_b2, *ln2_s, *ln2_b;
  const float *we_b, *pe_b1, *pe_b2, *wr_b, *pr_b1, *pr_b2;
  const float *ln_e_s, *ln_e_b, *ln_r_s, *ln_r_b;
  const float *eq, *ek, *ev;
  float* out;
};

__global__ void __launch_bounds__(THREADS)
pair_layer_kernel(KParams P)
{
  extern __shared__ char smem[];
  float* sPair = (float*)(smem + SM_PAIR);          // residual stream f32
  u16*   sAbf  = (u16*)(smem + SM_ABF);             // bf16 A operand
  u16*   sQ    = (u16*)(smem + SM_R3);              // 96x256 bf16
  u16*   sK    = sQ + ROWS * 256;
  u16*   sV    = sK + ROWS * 256;
  u16*   sIn   = sQ;                                // 96x512 bf16 (spans Q+K)
  float* msgE  = (float*)(smem + SM_R3 + 49152);    // f32 96x256 (spans K+V)
  float* sEq   = (float*)(smem + SM_MISC);
  float* sEk   = sEq + 160;
  float* sEv   = sEk + 160;
  int*   sValid = (int*)(sEv + 160);                // [b][p]

  const int tid = threadIdx.x;
  const int wave = tid >> 5, lane = tid & 31;
  const int b0 = blockIdx.x * BT;

  // ---- init: bias tables + validity ----
  for (int i = tid; i < 160; i += THREADS) { sEq[i] = P.eq[i]; sEk[i] = P.ek[i]; sEv[i] = P.ev[i]; }
  if (tid < BT * 6) {
    int b = tid / 6, p = tid % 6;
    int e = 2 * p;
    int rl = (p == 0) ? 1 : (2 * p - 1);
    const unsigned char* mrow = P.mask + (size_t)(b0 + b) * 121;
    sValid[tid] = (mrow[rl * 12] && mrow[e * 12]) ? 1 : 0;
  }
  __syncthreads();

  // ---- gather (ent || rel) * valid -> sIn bf16 96x512 ----
  for (int i = tid; i < ROWS * 512; i += THREADS) {
    int row = i >> 9, col = i & 511;
    int p = row >> 4, b = row & 15;
    float val = 0.f;
    if (sValid[b * 6 + p]) {
      int tokl = (col < 256) ? (2 * p) : ((p == 0) ? 1 : (2 * p - 1));
      int c = col & 255;
      val = P.x[((size_t)(b0 + b) * 11 + tokl) * 256 + c];
    }
    sIn[i] = f2bf(val);
  }
  __syncthreads();

  // ---- pair = gelu([ent||rel] @ projW + b) ----
  gemm(sIn, 512, P.ws + OFF_PROJ, 512, P.proj_b, nullptr, nullptr,
       sPair, false, sAbf, /*gelu*/1, 6, 16);
  __syncthreads();

  // ---- transformer layers ----
  for (int li = 0; li < 2; li++) {
    const long wl = (long)li * 65536;
    gemm(sAbf, 256, P.ws + OFF_WQ + wl, 256, P.bq + li * 256, nullptr, nullptr,
         nullptr, false, sQ, 0, 6, 8);
    gemm(sAbf, 256, P.ws + OFF_WK + wl, 256, P.bk + li * 256, nullptr, nullptr,
         nullptr, false, sK, 0, 6, 8);
    gemm(sAbf, 256, P.ws + OFF_WV + wl, 256, P.bv + li * 256, nullptr, nullptr,
         nullptr, false, sV, 0, 6, 8);
    __syncthreads();

    // ---- attention: one thread per (batch, head); o overwrites sQ in place ----
    if (tid < BT * 8) {
      int b = tid >> 3, h = tid & 7;
      int cb = h * 32;
      int labm[6][6];
      for (int i = 0; i < 6; i++) {
        int vi = sValid[b * 6 + i];
        for (int j = 0; j < 6; j++) {
          int vj = sValid[b * 6 + j];
          int lab = (vi && vj) ? c_lab[i * 6 + j] : 0;
          if (i == j && vi) lab = 1;
          labm[i][j] = lab;
        }
      }
      float a[6][6];
      for (int i = 0; i < 6; i++) {
        float qi[32];
        for (int d = 0; d < 32; d++) qi[d] = bf2f(sQ[((i << 4) + b) * 256 + cb + d]);
        for (int j = 0; j < 6; j++) {
          int lab = labm[i][j];
          if (lab > 0) {
            float s1 = 0.f, s2 = 0.f, s3 = 0.f;
            for (int d = 0; d < 32; d++) {
              float kj = bf2f(sK[((j << 4) + b) * 256 + cb + d]);
              s1 += qi[d] * kj;
              s2 += qi[d] * sEk[lab * 32 + d];
              s3 += kj * sEq[lab * 32 + d];
            }
            a[i][j] = (s1 + s2 + s3) * 0.17677669529663687f;
          } else a[i][j] = -1e9f;
        }
      }
      for (int i = 0; i < 6; i++) {
        float mx = a[i][0];
        for (int j = 1; j < 6; j++) mx = fmaxf(mx, a[i][j]);
        float ss = 0.f;
        for (int j = 0; j < 6; j++) { a[i][j] = expf(a[i][j] - mx); ss += a[i][j]; }
        float inv = 1.f / ss;
        for (int j = 0; j < 6; j++) a[i][j] *= inv;
      }
      for (int i = 0; i < 6; i++) {
        for (int d = 0; d < 32; d++) {
          float acc = 0.f;
          for (int j = 0; j < 6; j++) {
            int lab = labm[i][j];
            acc += a[i][j] * (bf2f(sV[((j << 4) + b) * 256 + cb + d]) + sEv[lab * 32 + d]);
          }
          sQ[((i << 4) + b) * 256 + cb + d] = f2bf(acc);
        }
      }
    }
    __syncthreads();

    // o @ wo + bo, accumulated onto residual stream (pair += ...)
    gemm(sQ, 256, P.ws + OFF_WO + wl, 256, P.bo + li * 256, nullptr, nullptr,
         sPair, true, nullptr, 0, 6, 8);
    __syncthreads();
    ln_rows(sPair, sAbf, P.ln1_s + li * 256, P.ln1_b + li * 256, nullptr);  // hmid
    __syncthreads();

    // FFN 256->1024->256, chunked by 256 over the hidden dim
    for (int c = 0; c < 4; c++) {
      gemm(sAbf, 256, P.ws + OFF_F1 + (long)li * 262144 + (long)c * 65536, 256,
           P.ffn_b1 + li * 1024 + c * 256, nullptr, nullptr,
           nullptr, false, sQ /*sFbf*/, /*gelu*/1, 6, 8);
      __syncthreads();
      gemm(sQ, 256, P.ws + OFF_F2 + (long)li * 262144 + c * 256, 1024,
           (c == 0) ? (P.ffn_b2 + li * 256) : nullptr, nullptr, nullptr,
           sPair, true, nullptr, 0, 6, 8);   // sPair = hmid + f
      __syncthreads();
    }
    ln_rows(sPair, sAbf, P.ln2_s + li * 256, P.ln2_b + li * 256, sValid);  // * valid
    __syncthreads();
  }

  // ---- role MLP chain (entities): pair -> we -> relu(pe1) -> pe2 -> msgE ----
  gemm(sAbf, 256, P.ws + OFF_WE, 256, P.we_b, c_roleB, c_roleBias,
       nullptr, false, sK, 0, 6, 8);
  __syncthreads();
  gemm(sK, 256, P.ws + OFF_PE1, 256, P.pe_b1, c_roleB, c_roleBias,
       nullptr, false, sQ, /*relu*/2, 6, 8);
  __syncthreads();
  gemm(sQ, 256, P.ws + OFF_PE2, 256, P.pe_b2, c_roleB, c_roleBias,
       msgE, false, nullptr, 0, 6, 8);
  __syncthreads();

  // ---- role MLP chain (relations): msgR lands in sPair (pair f32 is dead) ----
  gemm(sAbf, 256, P.ws + OFF_WR, 256, P.wr_b, c_roleB, c_roleBias,
       nullptr, false, sQ, 0, 6, 8);
  __syncthreads();
  gemm(sQ, 256, P.ws + OFF_PR1, 256, P.pr_b1, c_roleB, c_roleBias,
       nullptr, false, sAbf, /*relu*/2, 6, 8);
  __syncthreads();
  float* msgR = sPair;
  gemm(sAbf, 256, P.ws + OFF_PR2, 256, P.pr_b2, c_roleB, c_roleBias,
       msgR, false, nullptr, 0, 6, 8);
  __syncthreads();

  // ---- scatter + two LayerNorm stages, per (batch, token) row ----
  for (int r = wave; r < BT * 11; r += 8) {
    int b = r / 11, l = r % 11;
    const float* xrow = P.x + ((size_t)(b0 + b) * 11 + l) * 256;
    float* orow = P.out + ((size_t)(b0 + b) * 11 + l) * 256;
    float t[8]; float sum = 0.f, sq = 0.f;
    for (int k = 0; k < 8; k++) {
      int c = lane + 32 * k;
      float me = 0.f;
      if ((l & 1) == 0) me = msgE[(((l >> 1) << 4) + b) * 256 + c];
      float v = xrow[c] + gelu_f(me);
      t[k] = v; sum += v; sq += v * v;
    }
    for (int m = 16; m >= 1; m >>= 1) { sum += __shfl_xor(sum, m, 32); sq += __shfl_xor(sq, m, 32); }
    float mu = sum * (1.f / 256.f);
    float rs = rsqrtf(sq * (1.f / 256.f) - mu * mu + 1e-5f);
    float sum2 = 0.f, sq2 = 0.f;
    for (int k = 0; k < 8; k++) {
      int c = lane + 32 * k;
      float v = (t[k] - mu) * rs * P.ln_e_s[c] + P.ln_e_b[c];
      float mr = 0.f;
      if (l == 1) mr = 0.5f * (msgR[b * 256 + c] + msgR[(16 + b) * 256 + c]);
      else if (l & 1) mr = msgR[((((l + 1) >> 1) << 4) + b) * 256 + c];
      v = v + gelu_f(mr);
      t[k] = v; sum2 += v; sq2 += v * v;
    }
    for (int m = 16; m >= 1; m >>= 1) { sum2 += __shfl_xor(sum2, m, 32); sq2 += __shfl_xor(sq2, m, 32); }
    float mu2 = sum2 * (1.f / 256.f);
    float rs2 = rsqrtf(sq2 * (1.f / 256.f) - mu2 * mu2 + 1e-5f);
    for (int k = 0; k < 8; k++) {
      int c = lane + 32 * k;
      orow[c] = (t[k] - mu2) * rs2 * P.ln_r_s[c] + P.ln_r_b[c];
    }
  }
}

// src: [batch][K][N] f32 -> dst: [batch][N][K] bf16
__global__ void transpose_bf16_kernel(const float* src, u16* dst, int K, int N, int batch)
{
  long total = (long)batch * K * N;
  long kn = (long)K * N;
  for (long i = (long)blockIdx.x * blockDim.x + threadIdx.x; i < total;
       i += (long)gridDim.x * blockDim.x) {
    long b = i / kn; long rem = i - b * kn;
    int n = (int)(rem / K); int k = (int)(rem - (long)n * K);
    dst[i] = f2bf(src[b * kn + (long)k * N + n]);
  }
}

extern "C" void kernel_launch(void* const* d_in, const int* in_sizes, int n_in,
                              void* d_out, int out_size, void* d_ws, size_t ws_size,
                              hipStream_t stream) {
  (void)in_sizes; (void)n_in; (void)out_size; (void)ws_size;
  u16* ws = (u16*)d_ws;

  struct TD { int idx; long off; int K, N, batch; };
  const TD tds[] = {
    {3,  OFF_PROJ, 512, 256, 1},
    {5,  OFF_WQ,   256, 256, 2},
    {7,  OFF_WK,   256, 256, 2},
    {9,  OFF_WV,   256, 256, 2},
    {11, OFF_WO,   256, 256, 2},
    {15, OFF_F1,   256, 1024, 2},
    {17, OFF_F2,   1024, 256, 2},
    {21, OFF_WE,   256, 256, 3},
    {23, OFF_PE1,  256, 256, 3},
    {25, OFF_PE2,  256, 256, 3},
    {27, OFF_WR,   256, 256, 3},
    {29, OFF_PR1,  256, 256, 3},
    {31, OFF_PR2,  256, 256, 3},
  };
  for (int i = 0; i < (int)(sizeof(tds) / sizeof(tds[0])); i++) {
    long total = (long)tds[i].batch * tds[i].K * tds[i].N;
    int grid = (int)((total + 255) / 256);
    if (grid > 2048) grid = 2048;
    transpose_bf16_kernel<<<grid, 256, 0, stream>>>(
        (const float*)d_in[tds[i].idx], ws + tds[i].off, tds[i].K, tds[i].N, tds[i].batch);
  }

  KParams P;
  P.x      = (const float*)d_in[0];
  P.mask   = (const unsigned char*)d_in[1];
  P.ws     = ws;
  P.proj_b = (const float*)d_in[4];
  P.bq     = (const float*)d_in[6];
  P.bk     = (const float*)d_in[8];
  P.bv     = (const float*)d_in[10];
  P.bo     = (const float*)d_in[12];
  P.ln1_s  = (const float*)d_in[13];
  P.ln1_b  = (const float*)d_in[14];
  P.ffn_b1 = (const float*)d_in[16];
  P.ffn_b2 = (const float*)d_in[18];
  P.ln2_s  = (const float*)d_in[19];
  P.ln2_b  = (const float*)d_in[20];
  P.we_b   = (const float*)d_in[22];
  P.pe_b1  = (const float*)d_in[24];
  P.pe_b2  = (const float*)d_in[26];
  P.wr_b   = (const float*)d_in[28];
  P.pr_b1  = (const float*)d_in[30];
  P.pr_b2  = (const float*)d_in[32];
  P.ln_e_s = (const float*)d_in[33];
  P.ln_e_b = (const float*)d_in[34];
  P.ln_r_s = (const float*)d_in[35];
  P.ln_r_b = (const float*)d_in[36];
  P.eq     = (const float*)d_in[37];
  P.ek     = (const float*)d_in[38];
  P.ev     = (const float*)d_in[39];
  P.out    = (float*)d_out;

  const int nblocks = 16384 / BT;  // 1024
  pair_layer_kernel<<<nblocks, THREADS, SM_TOTAL, stream>>>(P);
}
